// SOMLayer_42442866819831
// MI455X (gfx1250) — matrix operand order repaired and checked by reading, not compiled, not run on hardware
//
#include <hip/hip_runtime.h>

#define DIM   512
#define BATCH 8192
#define NMAP  1024

// 1 / (2 * (35.2 * 0.08)^2) = 1 / 15.859712
#define INV2S2 0.0630528846f

typedef __attribute__((ext_vector_type(16))) __bf16       bf16x16;
typedef __attribute__((ext_vector_type(8)))  float        floatx8;
typedef __attribute__((ext_vector_type(4)))  float        floatx4;
typedef __attribute__((ext_vector_type(2)))  float        floatx2;
typedef __attribute__((ext_vector_type(4)))  unsigned int uintx4;
typedef __attribute__((ext_vector_type(2)))  unsigned int uintx2;

union FragU { uintx4 u[2]; bf16x16 v; };

__device__ __forceinline__ unsigned short f2bf(float f) {
  unsigned u = __float_as_uint(f);
  u += 0x7fffu + ((u >> 16) & 1u);      // round-to-nearest-even
  return (unsigned short)(u >> 16);
}

// CDNA5 async DMA global->LDS (16B per lane). dest LDS offset = generic
// address truncated to 32 bits (ISA 10.2: LDS aperture maps addr[31:0]).
__device__ __forceinline__ void async_copy_b128(void* lds, const void* gmem) {
  unsigned l = (unsigned)(unsigned long long)lds;
  asm volatile("global_load_async_to_lds_b128 %0, %1, off"
               :: "v"(l), "v"(gmem) : "memory");
}
__device__ __forceinline__ void async_wait0() {
  asm volatile("s_wait_asynccnt 0x0" ::: "memory");
}

// ---------------- x (f32) -> xb (bf16), row-major [b][d] ----------------
__global__ void k_cvt_x(const float* __restrict__ x, unsigned short* __restrict__ xb) {
  int i = blockIdx.x * 256 + threadIdx.x;          // float4 index
  floatx4 f = ((const floatx4*)x)[i];
  uintx2 p;
  p[0] = (unsigned)f2bf(f[0]) | ((unsigned)f2bf(f[1]) << 16);
  p[1] = (unsigned)f2bf(f[2]) | ((unsigned)f2bf(f[3]) << 16);
  ((uintx2*)xb)[i] = p;
}

// ------------- map (f32) -> mb (bf16) + m2[n] = ||map[n]||^2 ------------
__global__ void k_cvt_map(const float* __restrict__ m, unsigned short* __restrict__ mb,
                          float* __restrict__ m2) {
  __shared__ float red[256];
  int n = blockIdx.x, t = threadIdx.x;
  floatx2 f = ((const floatx2*)(m + n * DIM))[t];
  ((unsigned*)(mb + n * DIM))[t] = (unsigned)f2bf(f[0]) | ((unsigned)f2bf(f[1]) << 16);
  red[t] = f[0] * f[0] + f[1] * f[1];
  __syncthreads();
  for (int s = 128; s > 0; s >>= 1) { if (t < s) red[t] += red[t + s]; __syncthreads(); }
  if (t == 0) m2[n] = red[0];
}

// ---------------- xb [b][d] -> xbT [d][b] (64x64 LDS tiles) -------------
__global__ void k_transpose(const unsigned short* __restrict__ xb,
                            unsigned short* __restrict__ xbT) {
  __shared__ unsigned short tile[64][64];
  int t = threadIdx.x;                              // 256 threads
  int b0 = blockIdx.x * 64, d0 = blockIdx.y * 64;
#pragma unroll
  for (int j = 0; j < 2; j++) {                     // 512 b128 chunks / 256 thr
    int i = j * 256 + t, r = i >> 3, c8 = (i & 7) * 8;
    async_copy_b128(&tile[r][c8], xb + (b0 + r) * DIM + d0 + c8);
  }
  async_wait0();
  __syncthreads();
#pragma unroll
  for (int j = 0; j < 2; j++) {
    int i = j * 256 + t, r = i >> 3, c8 = (i & 7) * 8;   // r = d-row
    uintx4 v;
#pragma unroll
    for (int e = 0; e < 4; e++) {
      unsigned lo = tile[c8 + 2 * e][r], hi = tile[c8 + 2 * e + 1][r];
      v[e] = lo | (hi << 16);
    }
    *(uintx4*)(xbT + (d0 + r) * BATCH + b0 + c8) = v;
  }
}

__global__ void k_zeroS(float* __restrict__ S) {
  int i = blockIdx.x * 256 + threadIdx.x;
  if (i < NMAP) S[i] = 0.0f;
}

// ---- GEMM1 + argmin: scores[b,n] = m2[n] - 2 * (x[b] . map[n]) ---------
// 64 b-rows/WG, all 1024 n, K-chunks of 64, double-buffered async DMA.
__global__ void __launch_bounds__(128)
k_bmu(const unsigned short* __restrict__ xb, const unsigned short* __restrict__ mb,
      const float* __restrict__ m2, int* __restrict__ bmu) {
  __shared__ unsigned short xs[2][64][64];   // 16 KB
  __shared__ unsigned short ms[2][64][64];   // 16 KB
  const int t = threadIdx.x;
  const int w = t >> 5, l = t & 31, lh = l >> 4, ln = l & 15;
  const int b0 = blockIdx.x * 64;
  const int r0 = t >> 3, c8 = (t & 7) * 8;   // staging: 4 rows/thread, 16B chunks

  // iteration it = nc*8 + kc  (nc: 64-wide n-chunk, kc: 64-wide K-chunk)
  auto issue = [&](int buf, int it) {
    int nc_ = it >> 3, kc_ = it & 7;
#pragma unroll
    for (int j = 0; j < 4; j++) {
      int r = r0 + j * 16;
      async_copy_b128(&xs[buf][r][c8], xb + (b0 + r) * DIM + kc_ * 64 + c8);
      async_copy_b128(&ms[buf][r][c8], mb + (nc_ * 64 + r) * DIM + kc_ * 64 + c8);
    }
  };

  float rmin[8]; int ridx[8];
#pragma unroll
  for (int v = 0; v < 8; v++) { rmin[v] = 3.4e38f; ridx[v] = 0; }

  issue(0, 0);
  async_wait0();
  __syncthreads();

  for (int nc = 0; nc < NMAP / 64; nc++) {
    floatx8 acc[4] = {};                          // clean per-n-chunk zero init
    for (int kc = 0; kc < 8; kc++) {
      const int it = nc * 8 + kc;
      const int cur = it & 1;
      if (it + 1 < 128) issue(cur ^ 1, it + 1);   // DMA next chunk, overlap WMMA
#pragma unroll
      for (int ki = 0; ki < 2; ki++) {
        const int ko = ki * 32;
        FragU a;                                  // A: row M = b, K-contig
        a.u[0] = *(const uintx4*)&xs[cur][w * 16 + ln][ko + lh * 8];
        a.u[1] = *(const uintx4*)&xs[cur][w * 16 + ln][ko + lh * 8 + 16];
        FragU bf[4];                              // preload 4 B frags, 1 wait
#pragma unroll
        for (int nt = 0; nt < 4; nt++) {
          bf[nt].u[0] = *(const uintx4*)&ms[cur][nt * 16 + ln][ko + lh * 8];
          bf[nt].u[1] = *(const uintx4*)&ms[cur][nt * 16 + ln][ko + lh * 8 + 16];
        }
#pragma unroll
        for (int nt = 0; nt < 4; nt++)
          acc[nt] = __builtin_amdgcn_wmma_f32_16x16x32_bf16(
              false, a.v, false, bf[nt].v, (short)0, acc[nt], false, false);
      }
      async_wait0();
      __syncthreads();
    }
    // fold this 64-wide n-chunk into the running argmin
#pragma unroll
    for (int nt = 0; nt < 4; nt++) {
      int n = nc * 64 + nt * 16 + ln;
      float m2n = m2[n];
#pragma unroll
      for (int v = 0; v < 8; v++) {
        float s = m2n - 2.0f * acc[nt][v];
        if (s < rmin[v]) { rmin[v] = s; ridx[v] = n; }
      }
    }
  }

  // min-reduce over the 16 lanes of each half (row M = v + 8*lh lives there)
#pragma unroll
  for (int off = 8; off >= 1; off >>= 1) {
#pragma unroll
    for (int v = 0; v < 8; v++) {
      float om = __shfl_xor(rmin[v], off, 32);
      int   oi = __shfl_xor(ridx[v], off, 32);
      if (om < rmin[v] || (om == rmin[v] && oi < ridx[v])) { rmin[v] = om; ridx[v] = oi; }
    }
  }
  if (ln == 0) {
#pragma unroll
    for (int v = 0; v < 8; v++) bmu[b0 + w * 16 + lh * 8 + v] = ridx[v];
  }
}

// ------ lrT[n][b] = bf16(0.5*exp(-d2/15.86)),  S[n] += sum_b lr ---------
__global__ void k_lr(const int* __restrict__ bmu, unsigned short* __restrict__ lrT,
                     float* __restrict__ S) {
  __shared__ float red[256];
  int t = threadIdx.x;
  int n = blockIdx.y;
  int b = blockIdx.x * 256 + t;
  int code = bmu[b];
  int di = (n >> 5) - (code >> 5);
  int dj = (n & 31) - (code & 31);
  float lr = 0.5f * __expf(-(float)(di * di + dj * dj) * INV2S2);
  lrT[n * BATCH + b] = f2bf(lr);
  red[t] = lr;
  __syncthreads();
  for (int s = 128; s > 0; s >>= 1) { if (t < s) red[t] += red[t + s]; __syncthreads(); }
  if (t == 0) atomicAdd(&S[n], red[0]);
}

// ---- GEMM2: out[n,d] = map[n,d]*(1-S[n]) + sum_b lrT[n,b]*x[b,d] -------
__global__ void __launch_bounds__(128)
k_gemm2(const unsigned short* __restrict__ lrT, const unsigned short* __restrict__ xbT,
        const float* __restrict__ map, const float* __restrict__ S,
        float* __restrict__ out) {
  __shared__ unsigned short lrs[2][64][64];  // A tile: [n][b]
  __shared__ unsigned short xts[2][64][64];  // B tile: [d][b]
  const int t = threadIdx.x;
  const int w = t >> 5, l = t & 31, lh = l >> 4, ln = l & 15;
  const int d0 = blockIdx.x * 64, n0 = blockIdx.y * 64;
  const int r0 = t >> 3, c8 = (t & 7) * 8;

  auto issue = [&](int buf, int kc) {
#pragma unroll
    for (int j = 0; j < 4; j++) {
      int r = r0 + j * 16;
      async_copy_b128(&lrs[buf][r][c8], lrT + (n0 + r) * BATCH + kc * 64 + c8);
      async_copy_b128(&xts[buf][r][c8], xbT + (d0 + r) * BATCH + kc * 64 + c8);
    }
  };

  floatx8 acc[4] = {};
  issue(0, 0);
  async_wait0();
  __syncthreads();

  for (int kc = 0; kc < BATCH / 64; kc++) {
    const int cur = kc & 1;
    if (kc + 1 < BATCH / 64) issue(cur ^ 1, kc + 1);
#pragma unroll
    for (int ki = 0; ki < 2; ki++) {
      const int ko = ki * 32;
      FragU a;
      a.u[0] = *(const uintx4*)&lrs[cur][w * 16 + ln][ko + lh * 8];
      a.u[1] = *(const uintx4*)&lrs[cur][w * 16 + ln][ko + lh * 8 + 16];
      FragU bf[4];
#pragma unroll
      for (int dt = 0; dt < 4; dt++) {
        bf[dt].u[0] = *(const uintx4*)&xts[cur][dt * 16 + ln][ko + lh * 8];
        bf[dt].u[1] = *(const uintx4*)&xts[cur][dt * 16 + ln][ko + lh * 8 + 16];
      }
#pragma unroll
      for (int dt = 0; dt < 4; dt++)
        acc[dt] = __builtin_amdgcn_wmma_f32_16x16x32_bf16(
            false, a.v, false, bf[dt].v, (short)0, acc[dt], false, false);
    }
    async_wait0();
    __syncthreads();
  }
#pragma unroll
  for (int dt = 0; dt < 4; dt++) {
    int d = d0 + dt * 16 + ln;
#pragma unroll
    for (int v = 0; v < 8; v++) {
      int n = n0 + w * 16 + lh * 8 + v;
      out[n * DIM + d] = map[n * DIM + d] * (1.0f - S[n]) + acc[dt][v];
    }
  }
}

extern "C" void kernel_launch(void* const* d_in, const int* in_sizes, int n_in,
                              void* d_out, int out_size, void* d_ws, size_t ws_size,
                              hipStream_t stream) {
  const float* x   = (const float*)d_in[0];
  const float* som = (const float*)d_in[1];
  // d_in[2] (locs) unused: grid is the regular 32x32 meshgrid, n = i*32 + j
  float* out = (float*)d_out;

  char* ws = (char*)d_ws;
  unsigned short* xb  = (unsigned short*)(ws + 0);          //  8 MB  x  bf16 [b][d]
  unsigned short* xbT = (unsigned short*)(ws + 8388608);    //  8 MB  x  bf16 [d][b]
  unsigned short* mb  = (unsigned short*)(ws + 16777216);   //  1 MB  map bf16
  unsigned short* lrT = (unsigned short*)(ws + 17825792);   // 16 MB  lr bf16 [n][b]
  float*          m2  = (float*)(ws + 34603008);            //  4 KB
  float*          S   = (float*)(ws + 34607104);            //  4 KB
  int*            bmu = (int*)(ws + 34611200);              // 32 KB

  k_cvt_x    <<<BATCH * DIM / 4 / 256, 256, 0, stream>>>(x, xb);
  k_cvt_map  <<<NMAP, 256, 0, stream>>>(som, mb, m2);
  k_transpose<<<dim3(BATCH / 64, DIM / 64), 256, 0, stream>>>(xb, xbT);
  k_zeroS    <<<NMAP / 256, 256, 0, stream>>>(S);
  k_bmu      <<<BATCH / 64, 128, 0, stream>>>(xb, mb, m2, bmu);
  k_lr       <<<dim3(BATCH / 256, NMAP), 256, 0, stream>>>(bmu, lrT, S);
  k_gemm2    <<<dim3(DIM / 64, NMAP / 64), 128, 0, stream>>>(lrT, xbT, som, S, out);
}
